// BatchedAdjacency_40793599377963
// MI455X (gfx1250) — matrix operand order repaired and checked by reading, not compiled, not run on hardware
//
#include <hip/hip_runtime.h>

typedef __attribute__((ext_vector_type(16))) _Float16 v16h;
typedef __attribute__((ext_vector_type(8)))  float    v8f;

#define BS     4
#define LCH    16          // signal channels (= WMMA N)
#define DG     5           // guide dims
#define NPIX   4096        // 64*64
#define WAVES  8           // waves per block (one 16-row tile each)
#define TJ     32          // j-chunk = WMMA K
#define NCHUNK (NPIX / TJ) // 128

__global__ __launch_bounds__(WAVES * 32)
void BatchedAdjacency_wmma_kernel(const float* __restrict__ src,
                                  const float* __restrict__ guide,
                                  float* __restrict__ out)
{
    // double-buffered chunk staging, shared by all 8 waves of the block
    __shared__ __align__(32) _Float16 lds_s[2][TJ * LCH]; // S chunk: 32 rows x 16 ch (f16)
    __shared__ float lds_r[2][TJ * 6];                    // r_j[0..4], [5] = -0.5*log2e*||r_j||^2

    const int tid  = threadIdx.x;
    const int lane = tid & 31;
    const int wave = tid >> 5;

    const int groups = NPIX / (16 * WAVES);               // 32 row-groups per batch
    const int b   = blockIdx.x / groups;
    const int grp = blockIdx.x % groups;
    const int i0  = grp * (16 * WAVES) + wave * 16;       // this wave's 16-row tile

    const float* srcb = src   + (size_t)b * LCH * NPIX;
    const float* gdb  = guide + (size_t)b * DG  * NPIX;

    // Per-lane row features (A layout: lane L -> row M = L%16; half = L/16),
    // pre-scaled by log2(e) so the inner loop feeds v_exp_f32 directly.
    const float LOG2E = 1.4426950408889634f;
    const int mrow = lane & 15;
    const int half = lane >> 4;
    const int irow = i0 + mrow;
    const float ri0 = gdb[0 * NPIX + irow];
    const float ri1 = gdb[1 * NPIX + irow];
    const float ri2 = gdb[2 * NPIX + irow];
    const float ri3 = gdb[3 * NPIX + irow];
    const float ri4 = gdb[4 * NPIX + irow];
    const float qi0 = ri0 * LOG2E;
    const float qi1 = ri1 * LOG2E;
    const float qi2 = ri2 * LOG2E;
    const float qi3 = ri3 * LOG2E;
    const float qi4 = ri4 * LOG2E;
    const float hsqi = -0.5f * (qi0*ri0 + qi1*ri1 + qi2*ri2 + qi3*ri3 + qi4*ri4);

    // staging assignment: thread -> (channel l0, row-in-chunk jj0) for S,
    // threads 0..31 additionally own one guide row each.
    const int l0  = tid >> 4;
    const int jj0 = tid & 15;
    const float* sPtr = srcb + (size_t)l0 * NPIX + jj0;   // + j0 (and +16 for upper half)
    const bool  gAct  = (tid < TJ);
    const float* gPtr = gdb + tid;                        // + d*NPIX + j0

    v8f c = {0.f, 0.f, 0.f, 0.f, 0.f, 0.f, 0.f, 0.f};

    // ---- prologue: stage chunk 0 into buffer 0 ----
    {
        const float s0 = sPtr[0];
        const float s1 = sPtr[16];
        lds_s[0][l0 * TJ + jj0]        = (_Float16)s0;
        lds_s[0][l0 * TJ + jj0 + 16]   = (_Float16)s1;
        if (gAct) {
            const float g0 = gPtr[0 * NPIX], g1 = gPtr[1 * NPIX], g2 = gPtr[2 * NPIX],
                        g3 = gPtr[3 * NPIX], g4 = gPtr[4 * NPIX];
            float* r = lds_r[0] + tid * 6;
            r[0] = g0; r[1] = g1; r[2] = g2; r[3] = g3; r[4] = g4;
            r[5] = -0.5f * LOG2E * (g0*g0 + g1*g1 + g2*g2 + g3*g3 + g4*g4);
        }
    }
    __syncthreads();

    int buf = 0;
    for (int t = 0; t < NCHUNK; ++t) {
        // ---- issue next-chunk global loads NOW (latency hidden by compute).
        // Last iteration: clamp index and commit to the dead buffer (branch-free).
        const int jn = (t + 1 < NCHUNK ? t + 1 : t) * TJ;
        const float n_s0 = sPtr[jn];
        const float n_s1 = sPtr[jn + 16];
        float n_g0 = 0.f, n_g1 = 0.f, n_g2 = 0.f, n_g3 = 0.f, n_g4 = 0.f;
        if (gAct) {
            n_g0 = gPtr[0 * NPIX + jn]; n_g1 = gPtr[1 * NPIX + jn];
            n_g2 = gPtr[2 * NPIX + jn]; n_g3 = gPtr[3 * NPIX + jn];
            n_g4 = gPtr[4 * NPIX + jn];
        }

        // ---- build A tile (16x32 f16 Gaussian weights) in WMMA A layout ----
        // lane L: VGPR v holds K = {k0, k0+1}, k0 = 16*(L/16) + 2*v
        union { v16h v; _Float16 h[16]; } au;
        #pragma unroll
        for (int v = 0; v < 8; ++v) {
            const int k0 = (half << 4) + (v << 1);
            #pragma unroll
            for (int p = 0; p < 2; ++p) {
                const float* rj = lds_r[buf] + (k0 + p) * 6;
                float dot = qi0 * rj[0];
                dot = fmaf(qi1, rj[1], dot);
                dot = fmaf(qi2, rj[2], dot);
                dot = fmaf(qi3, rj[3], dot);
                dot = fmaf(qi4, rj[4], dot);
                // log2e*(dot - 0.5|ri|^2 - 0.5|rj|^2), clamped to <= 0
                const float arg = fminf(dot + hsqi + rj[5], 0.0f);
                au.h[2 * v + p] = (_Float16)__builtin_amdgcn_exp2f(arg);
            }
        }

        // ---- B operand: lane L = channel N=L%16, K = 16*(L/16)+0..15 of S chunk (f16) ----
        const v16h bm = *(const v16h*)(lds_s[buf] + mrow * TJ + (half << 4));

        // ---- C += A * B  (v_wmma_f32_16x16x32_f16) ----
        c = __builtin_amdgcn_wmma_f32_16x16x32_f16(
                false, au.v, false, bm, (short)0, c, false, false);

        // ---- commit prefetched chunk into the other buffer ----
        const int nb = buf ^ 1;
        lds_s[nb][l0 * TJ + jj0]        = (_Float16)n_s0;
        lds_s[nb][l0 * TJ + jj0 + 16]   = (_Float16)n_s1;
        if (gAct) {
            float* r = lds_r[nb] + tid * 6;
            r[0] = n_g0; r[1] = n_g1; r[2] = n_g2; r[3] = n_g3; r[4] = n_g4;
            r[5] = -0.5f * LOG2E *
                   (n_g0*n_g0 + n_g1*n_g1 + n_g2*n_g2 + n_g3*n_g3 + n_g4*n_g4);
        }
        __syncthreads();   // single barrier per chunk
        buf = nb;
    }

    // ---- epilogue: C layout lane L -> (M = v + 8*half, N = L%16) ----
    // lane's channel l = lane%16; pixels i = i0 + 8*half + v (consecutive in v)
    const int l     = lane & 15;
    const int ibase = i0 + (half << 3);
    const float* sp = srcb + (size_t)l * NPIX + ibase;
    float* op = out + (size_t)b * LCH * NPIX + (size_t)l * NPIX + ibase;

    const float4 s0 = *(const float4*)(sp);
    const float4 s1 = *(const float4*)(sp + 4);
    float4 r0, r1;
    r0.x = c[0] - s0.x; r0.y = c[1] - s0.y; r0.z = c[2] - s0.z; r0.w = c[3] - s0.w;
    r1.x = c[4] - s1.x; r1.y = c[5] - s1.y; r1.z = c[6] - s1.z; r1.w = c[7] - s1.w;
    *(float4*)(op)     = r0;
    *(float4*)(op + 4) = r1;
}

extern "C" void kernel_launch(void* const* d_in, const int* in_sizes, int n_in,
                              void* d_out, int out_size, void* d_ws, size_t ws_size,
                              hipStream_t stream)
{
    (void)in_sizes; (void)n_in; (void)out_size; (void)d_ws; (void)ws_size;
    const float* src   = (const float*)d_in[0];   // [BS, L, H, W] f32
    const float* guide = (const float*)d_in[1];   // [BS, D, H, W] f32
    float* out = (float*)d_out;                   // [BS, L, H, W] f32

    const int groups = NPIX / (16 * WAVES);       // 32
    dim3 grid(BS * groups);                       // 128 blocks
    dim3 block(WAVES * 32);                       // 256 threads = 8 waves
    BatchedAdjacency_wmma_kernel<<<grid, block, 0, stream>>>(src, guide, out);
}
